// Llama_48189533061802
// MI455X (gfx1250) — matrix operand order repaired and checked by reading, not compiled, not run on hardware
//
#include <hip/hip_runtime.h>

// ---------------------------------------------------------------------------
// Model constants
// ---------------------------------------------------------------------------
#define DIM    1024
#define NL     2
#define NH     16
#define HD     64
#define VOCAB  32000
#define NE     8
#define TOPK   2
#define HID    1408
#define SEQ    2048
#define EPSF   1e-5f
#define THETAF 10000.0f

// Grouped-MoE row space: SEQ*TOPK real rows, padded per expert to 64-row tiles
#define MAXR      (SEQ*TOPK + 64*NE)   // 4608
#define MAX_TILES (MAXR/64)            // 72

// ---------------------------------------------------------------------------
// bf16 helpers (round-to-nearest-even), fragment types for CDNA5 WMMA
// ---------------------------------------------------------------------------
typedef __attribute__((ext_vector_type(16))) __bf16 v16bf;
typedef __attribute__((ext_vector_type(8)))  float  v8f;

union Frag16 { v16bf v; unsigned int u[8]; };

__device__ inline unsigned short f2bf(float f) {
  unsigned int u = __float_as_uint(f);
  unsigned int r = u + 0x7FFFu + ((u >> 16) & 1u);
  return (unsigned short)(r >> 16);
}
__device__ inline float bf2f(unsigned short h) {
  return __uint_as_float(((unsigned int)h) << 16);
}
__device__ inline v8f v8f_zero() {
  v8f z;
#pragma unroll
  for (int i = 0; i < 8; i++) z[i] = 0.f;
  return z;
}
__device__ inline v8f wmma_bf16(const Frag16& a, const Frag16& b, v8f c) {
  // D = A(16x32 bf16) * B(32x16 bf16) + C(16x16 f32)
  return __builtin_amdgcn_wmma_f32_16x16x32_bf16(
      false, a.v, false, b.v, (short)0, c, false, false);
}

// ---------------------------------------------------------------------------
// CDNA5 async global->LDS staging (ASYNCcnt path), probed at compile time.
// Builtin signature (diagnostic-verified): (int4 as1*, int4 as3*, Iimm, Iimm).
// ---------------------------------------------------------------------------
#if defined(__AMDGCN__) && __has_builtin(__builtin_amdgcn_global_load_async_to_lds_b128)
#define USE_ASYNC_LDS 1
#else
#define USE_ASYNC_LDS 0
#endif

typedef int v4i __attribute__((vector_size(16)));
typedef __attribute__((address_space(1))) v4i v4i_g;   // global int4
typedef __attribute__((address_space(3))) v4i v4i_l;   // LDS int4

__device__ inline void wait_async_le1() {
#if defined(__AMDGCN__)
#if __has_builtin(__builtin_amdgcn_s_wait_asynccnt)
  __builtin_amdgcn_s_wait_asynccnt(1);
#else
  asm volatile("s_wait_asynccnt 0x1" ::: "memory");
#endif
#endif
}
__device__ inline void wait_async_le0() {
#if defined(__AMDGCN__)
#if __has_builtin(__builtin_amdgcn_s_wait_asynccnt)
  __builtin_amdgcn_s_wait_asynccnt(0);
#else
  asm volatile("s_wait_asynccnt 0x0" ::: "memory");
#endif
#endif
}

// ---------------------------------------------------------------------------
// Embedding gather: h[s,:] = tok_emb[tokens[s],:]
// ---------------------------------------------------------------------------
__global__ __launch_bounds__(256)
void embed_kernel(const int* __restrict__ tokens, const float* __restrict__ emb,
                  float* __restrict__ h) {
  int s = blockIdx.x;
  long long tok = tokens[s];
  const float* src = emb + tok * DIM;
  float* dst = h + (long long)s * DIM;
  for (int d = threadIdx.x; d < DIM; d += 256) dst[d] = src[d];
}

// ---------------------------------------------------------------------------
// RMSNorm (fp32 in -> bf16 normalized out)
// ---------------------------------------------------------------------------
__global__ __launch_bounds__(256)
void rmsnorm_kernel(const float* __restrict__ x, const float* __restrict__ w,
                    unsigned short* __restrict__ out) {
  int s = blockIdx.x;
  __shared__ float red[8];
  const float* xr = x + (long long)s * DIM;
  float ss = 0.f;
  for (int d = threadIdx.x; d < DIM; d += 256) { float v = xr[d]; ss += v * v; }
#pragma unroll
  for (int off = 1; off < 32; off <<= 1) ss += __shfl_xor(ss, off, 32);
  if ((threadIdx.x & 31) == 0) red[threadIdx.x >> 5] = ss;
  __syncthreads();
  float tot = 0.f;
#pragma unroll
  for (int i = 0; i < 8; i++) tot += red[i];
  float r = rsqrtf(tot / (float)DIM + EPSF);
  unsigned short* o = out + (long long)s * DIM;
  for (int d = threadIdx.x; d < DIM; d += 256) o[d] = f2bf(xr[d] * r * w[d]);
}

// ---------------------------------------------------------------------------
// RoPE: fp32 q,k -> rotated bf16 q,k
// ---------------------------------------------------------------------------
__global__ __launch_bounds__(256)
void rope_kernel(const float* __restrict__ qf, const float* __restrict__ kf,
                 unsigned short* __restrict__ qb, unsigned short* __restrict__ kb,
                 const int* __restrict__ startPos) {
  int s = blockIdx.x;
  float t = (float)(s + startPos[0]);
  for (int p = threadIdx.x; p < NH * (HD / 2); p += 256) {
    int hh = p / (HD / 2);
    int d2 = p % (HD / 2);
    float inv = __powf(THETAF, -(float)(2 * d2) / (float)HD);
    float ang = t * inv;
    float cs = __cosf(ang), sn = __sinf(ang);
    long long base = (long long)s * DIM + hh * HD + 2 * d2;
    float a = qf[base], b = qf[base + 1];
    qb[base]     = f2bf(a * cs - b * sn);
    qb[base + 1] = f2bf(a * sn + b * cs);
    a = kf[base]; b = kf[base + 1];
    kb[base]     = f2bf(a * cs - b * sn);
    kb[base + 1] = f2bf(a * sn + b * cs);
  }
}

// ---------------------------------------------------------------------------
// Generic bf16-WMMA GEMM:
//   C[M,N](f32) (+= Resid) = gather(A_bf16)[M,Kd] x cvt_bf16(W_f32)[Kd,N]
// CONTRACT: M % 64 == 0, N % 128 == 0, Kd % 32 == 0 (all launches satisfy).
// Grouped-MoE: tileExpert[rowTile] selects weight slab W + e*wStride,
//              rowTok[] gathers A rows (negative => zero row).
// Block: 256 threads (8 waves), tile 64x128, BK=32, wave computes 32x32.
// A tile: double-buffered GLOBAL_LOAD_ASYNC_TO_LDS_B128 pipeline — tile ks+1
// DMA'd while tile ks computes; s_wait_asynccnt<=1 ensures residency (async
// loads complete in order). W tile: batched strided loads + fp32->bf16
// convert, transposed into LDS; global_prefetch of the next tile.
// ---------------------------------------------------------------------------
#define BM 64
#define BN 128
#define BK 32

__global__ __launch_bounds__(256)
void gemm_bf16_kernel(const unsigned short* __restrict__ A,
                      const float* __restrict__ W,
                      float* __restrict__ C, unsigned short* __restrict__ Cb,
                      const float* __restrict__ Resid,
                      int M, int N, int Kd,
                      const int* __restrict__ rowTok,
                      const int* __restrict__ tileExpert,
                      long long wStride) {
  int tileM = blockIdx.y;
  long long eoff = 0;
  if (tileExpert) {
    int e = tileExpert[tileM];
    if (e < 0) return;
    eoff = (long long)e * wStride;
  }
  const float* Wp = W + eoff;
  int row0 = tileM * BM;
  int col0 = blockIdx.x * BN;
  int tid  = threadIdx.x;
  int lane = tid & 31;
  int wave = tid >> 5;
  int wm = wave >> 2;   // 0..1 : 32-row sub-tile
  int wn = wave & 3;    // 0..3 : 32-col sub-tile

  __shared__ __align__(16) unsigned int As[2][BM][BK / 2];  // ping-pong A
  __shared__ __align__(16) unsigned int Ws[BN][BK / 2];     // transposed W

  v8f acc[2][2];
#pragma unroll
  for (int i = 0; i < 2; i++)
#pragma unroll
    for (int j = 0; j < 2; j++) acc[i][j] = v8f_zero();

  // ---- loop-invariant A-row resolution: one 16B chunk per thread ----
  int sRow  = tid >> 2;                 // 0..63
  int sQuad = tid & 3;                  // 0..3 (16B chunks of the 64B k-slice)
  const unsigned short* aSrc = nullptr; // chunk base (without kt)
  {
    int arow = row0 + sRow;
    bool ok = true;
    if (rowTok) { arow = rowTok[row0 + sRow]; ok = (arow >= 0); }
    if (ok) aSrc = A + (long long)arow * Kd + sQuad * 8;
  }
  if (!aSrc) {
    // rows that gather nothing stay zero in both buffers for the whole loop
    uint4 z = {0u, 0u, 0u, 0u};
    *(uint4*)&As[0][sRow][sQuad * 4] = z;
    *(uint4*)&As[1][sRow][sQuad * 4] = z;
  }

  // ---- loop-invariant W staging geometry: fixed column per thread ----
  int wCol  = tid & 127;                // 0..127
  int wRow0 = tid >> 7;                 // 0..1 (rows wRow0 + 2*it)
  const float* wColPtr = Wp + col0 + wCol;
  unsigned short* wDst = ((unsigned short*)Ws) + wCol * BK + wRow0;

  int mA  = lane & 15;
  int kbA = (lane < 16) ? 0 : 8;    // A fragment K base (ISA 16-bit A 16x32)
  int nB  = lane & 15;
  int kbB = (lane < 16) ? 0 : 8;    // B fragment K base

  int nsteps = Kd / BK;

#if USE_ASYNC_LDS
  // prime the pipeline: DMA tile 0 into buffer 0
  if (aSrc)
    __builtin_amdgcn_global_load_async_to_lds_b128(
        (v4i_g*)aSrc, (v4i_l*)&As[0][sRow][sQuad * 4], 0, 0);
#endif

  for (int ks = 0; ks < nsteps; ks++) {
    int kt = ks * BK;
    int cur = ks & 1;
#if USE_ASYNC_LDS
    // ---- issue DMA for tile ks+1 into the other buffer ----
    if (aSrc && (ks + 1 < nsteps))
      __builtin_amdgcn_global_load_async_to_lds_b128(
          (v4i_g*)(aSrc + kt + BK), (v4i_l*)&As[cur ^ 1][sRow][sQuad * 4], 0, 0);
#else
    if (aSrc)
      *(uint4*)&As[cur][sRow][sQuad * 4] = *(const uint4*)(aSrc + kt);
#endif
    // ---- stage W tile (32 x 128 f32 -> bf16, transposed): batched loads ----
    float wv[16];
#pragma unroll
    for (int it = 0; it < 16; it++)
      wv[it] = wColPtr[(long long)(kt + wRow0 + 2 * it) * N];
#pragma unroll
    for (int it = 0; it < 16; it++)
      wDst[2 * it] = f2bf(wv[it]);
    if (ks + 1 < nsteps)  // gfx1250 global_prefetch of next weight tile
      __builtin_prefetch(wColPtr + (long long)(kt + BK) * N, 0, 1);
#if USE_ASYNC_LDS
    // tile ks resident (async loads complete in order); ks+1 stays in flight
    if (ks + 1 < nsteps) wait_async_le1(); else wait_async_le0();
#endif
    __syncthreads();

    Frag16 af[2], bfr[2];
#pragma unroll
    for (int ms = 0; ms < 2; ms++) {
      int r = wm * 32 + ms * 16 + mA;
#pragma unroll
      for (int j = 0; j < 8; j++) {
        int kk = kbA + 2 * (j & 3) + ((j & 4) ? 16 : 0);
        af[ms].u[j] = As[cur][r][kk >> 1];
      }
    }
#pragma unroll
    for (int ns = 0; ns < 2; ns++) {
      int c = wn * 32 + ns * 16 + nB;
#pragma unroll
      for (int j = 0; j < 8; j++) bfr[ns].u[j] = Ws[c][(kbB >> 1) + (j & 3) + ((j & 4) ? 8 : 0)];
    }
#pragma unroll
    for (int ms = 0; ms < 2; ms++)
#pragma unroll
      for (int ns = 0; ns < 2; ns++)
        acc[ms][ns] = wmma_bf16(af[ms], bfr[ns], acc[ms][ns]);
    __syncthreads();
  }

  // ---- store (C fragment: VGPR r -> row r + 8*(lane>>4), col lane&15) ----
  int hi = lane >> 4;
  int nn = lane & 15;
#pragma unroll
  for (int ms = 0; ms < 2; ms++)
#pragma unroll
    for (int ns = 0; ns < 2; ns++)
#pragma unroll
      for (int r = 0; r < 8; r++) {
        int grow = row0 + wm * 32 + ms * 16 + r + 8 * hi;
        int gcol = col0 + wn * 32 + ns * 16 + nn;
        float v = acc[ms][ns][r];
        long long idx = (long long)grow * N + gcol;
        if (Resid) v += Resid[idx];
        if (C)  C[idx]  = v;
        if (Cb) Cb[idx] = f2bf(v);
      }
}

// ---------------------------------------------------------------------------
// Flash attention: one wave per (head, 16-query tile).
// Scores via WMMA (Q 16xHD  x  K^T), online softmax in C-fragment layout,
// P staged through LDS into A-fragment, P*V via WMMA. Output bf16.
// ---------------------------------------------------------------------------
__global__ __launch_bounds__(32)
void attn_kernel(const unsigned short* __restrict__ qb,
                 const unsigned short* __restrict__ kb,
                 const unsigned short* __restrict__ vb,
                 unsigned short* __restrict__ ob) {
  int q0 = blockIdx.x * 16;
  int h  = blockIdx.y;
  int lane = threadIdx.x;
  int hi = lane >> 4;
  int nn = lane & 15;

  __shared__ __align__(16) unsigned short Pl[16][32];

  // Q A-fragments for HD chunks [0,32) and [32,64)
  Frag16 aq[2];
  {
    int kbA = hi ? 8 : 0;
    const unsigned short* qrow = qb + (long long)(q0 + nn) * DIM + h * HD;
#pragma unroll
    for (int c = 0; c < 2; c++)
#pragma unroll
      for (int j = 0; j < 8; j++) {
        int kk = c * 32 + kbA + 2 * (j & 3) + ((j & 4) ? 16 : 0);
        aq[c].u[j] = *(const unsigned int*)(qrow + kk);
      }
  }

  v8f oacc[4];
#pragma unroll
  for (int c = 0; c < 4; c++) oacc[c] = v8f_zero();
  float mrow[8], lrow[8], pscale[8];
#pragma unroll
  for (int r = 0; r < 8; r++) { mrow[r] = -1e30f; lrow[r] = 0.f; }

  for (int k0 = 0; k0 <= q0 + 15; k0 += 32) {
    // ---- K^T B-fragments: [hd-chunk][key-subtile] ----
    Frag16 bk[2][2];
    int kbB = hi ? 8 : 0;
#pragma unroll
    for (int st = 0; st < 2; st++) {
      const unsigned short* krow = kb + (long long)(k0 + st * 16 + nn) * DIM + h * HD;
#pragma unroll
      for (int c = 0; c < 2; c++)
#pragma unroll
        for (int j = 0; j < 8; j++)
          bk[c][st].u[j] = *(const unsigned int*)(krow + c * 32 + kbB + 2 * (j & 3) + ((j & 4) ? 16 : 0));
    }
    v8f c0 = v8f_zero(), c1 = v8f_zero();
    c0 = wmma_bf16(aq[0], bk[0][0], c0);
    c0 = wmma_bf16(aq[1], bk[1][0], c0);
    c1 = wmma_bf16(aq[0], bk[0][1], c1);
    c1 = wmma_bf16(aq[1], bk[1][1], c1);

    // ---- causal mask + online softmax (row = r + 8*hi, col = k0(+16) + nn) ----
#pragma unroll
    for (int r = 0; r < 8; r++) {
      int rowq = q0 + r + 8 * hi;
      float s0 = c0[r] * 0.125f;
      float s1 = c1[r] * 0.125f;
      if (k0 + nn > rowq)      s0 = -1e9f;
      if (k0 + 16 + nn > rowq) s1 = -1e9f;
      float mx = fmaxf(s0, s1);
#pragma unroll
      for (int off = 1; off < 16; off <<= 1) mx = fmaxf(mx, __shfl_xor(mx, off, 32));
      float mnew = fmaxf(mrow[r], mx);
      float p0 = __expf(s0 - mnew);
      float p1 = __expf(s1 - mnew);
      float rs = p0 + p1;
#pragma unroll
      for (int off = 1; off < 16; off <<= 1) rs += __shfl_xor(rs, off, 32);
      float sc = __expf(mrow[r] - mnew);
      lrow[r] = lrow[r] * sc + rs;
      mrow[r] = mnew;
      pscale[r] = sc;
      Pl[r + 8 * hi][nn]      = f2bf(p0);
      Pl[r + 8 * hi][16 + nn] = f2bf(p1);
    }
    __syncthreads();

    // ---- P as A-fragment (16 queries x 32 keys) ----
    Frag16 pa;
    {
      int kbA = hi ? 8 : 0;
#pragma unroll
      for (int j = 0; j < 8; j++) {
        int kk = kbA + 2 * (j & 3) + ((j & 4) ? 16 : 0);
        pa.u[j] = *(const unsigned int*)&Pl[nn][kk];
      }
    }
    // ---- P*V: 4 hd chunks of 16 cols, K = 32 keys ----
#pragma unroll
    for (int c = 0; c < 4; c++) {
      Frag16 vbf;
#pragma unroll
      for (int j = 0; j < 8; j++) {
        int K = (hi ? 8 : 0) + 2 * (j & 3) + ((j & 4) ? 16 : 0);
        unsigned int lo = vb[(long long)(k0 + K) * DIM + h * HD + c * 16 + nn];
        unsigned int hi2 = vb[(long long)(k0 + K + 1) * DIM + h * HD + c * 16 + nn];
        vbf.u[j] = lo | (hi2 << 16);
      }
      v8f t = oacc[c];
#pragma unroll
      for (int r = 0; r < 8; r++) t[r] *= pscale[r];
      oacc[c] = wmma_bf16(pa, vbf, t);
    }
    __syncthreads();
  }

  // ---- normalize + store bf16 ----
#pragma unroll
  for (int c = 0; c < 4; c++)
#pragma unroll
    for (int r = 0; r < 8; r++) {
      float v = oacc[c][r] / lrow[r];
      ob[(long long)(q0 + r + 8 * hi) * DIM + h * HD + c * 16 + nn] = f2bf(v);
    }
}

// ---------------------------------------------------------------------------
// MoE router: per-token logits (fp32), softmax over 8, top-2 (no renorm),
// atomic per-expert counts. One wave per token.
// ---------------------------------------------------------------------------
__global__ __launch_bounds__(32)
void router_kernel(const unsigned short* __restrict__ xb,
                   const float* __restrict__ rw,
                   int* __restrict__ topIdx, float* __restrict__ topW,
                   int* __restrict__ cnt) {
  int t = blockIdx.x;
  int lane = threadIdx.x;
  float acc[NE];
#pragma unroll
  for (int e = 0; e < NE; e++) acc[e] = 0.f;
  for (int d = lane; d < DIM; d += 32) {
    float x = bf2f(xb[(long long)t * DIM + d]);
#pragma unroll
    for (int e = 0; e < NE; e++) acc[e] += x * rw[(long long)d * NE + e];
  }
#pragma unroll
  for (int e = 0; e < NE; e++)
#pragma unroll
    for (int off = 1; off < 32; off <<= 1) acc[e] += __shfl_xor(acc[e], off, 32);
  if (lane == 0) {
    float mx = acc[0];
#pragma unroll
    for (int e = 1; e < NE; e++) mx = fmaxf(mx, acc[e]);
    float p[NE], sum = 0.f;
#pragma unroll
    for (int e = 0; e < NE; e++) { p[e] = __expf(acc[e] - mx); sum += p[e]; }
#pragma unroll
    for (int e = 0; e < NE; e++) p[e] /= sum;
    int i1 = 0;
#pragma unroll
    for (int e = 1; e < NE; e++) if (p[e] > p[i1]) i1 = e;
    int i2 = (i1 == 0) ? 1 : 0;
#pragma unroll
    for (int e = 0; e < NE; e++) if (e != i1 && p[e] > p[i2]) i2 = e;
    topIdx[t * 2]     = i1; topW[t * 2]     = p[i1];
    topIdx[t * 2 + 1] = i2; topW[t * 2 + 1] = p[i2];
    atomicAdd(&cnt[i1], 1);
    atomicAdd(&cnt[i2], 1);
  }
}

__global__ __launch_bounds__(256)
void moe_init_kernel(int* __restrict__ cnt, int* __restrict__ tileExpert,
                     int* __restrict__ rowTok) {
  int i = blockIdx.x * 256 + threadIdx.x;
  if (i < NE) cnt[i] = 0;
  if (i < MAX_TILES) tileExpert[i] = -1;
  if (i < MAXR) rowTok[i] = -1;
}

__global__
void moe_offsets_kernel(const int* __restrict__ cnt, int* __restrict__ fill,
                        int* __restrict__ tileExpert) {
  if (blockIdx.x == 0 && threadIdx.x == 0) {
    int off = 0;
    for (int e = 0; e < NE; e++) {
      fill[e] = off;
      int cap = ((cnt[e] + 63) / 64) * 64;
      for (int tt = 0; tt < cap / 64; tt++) tileExpert[off / 64 + tt] = e;
      off += cap;
    }
  }
}

__global__ __launch_bounds__(256)
void moe_scatter_kernel(const int* __restrict__ topIdx, const float* __restrict__ topW,
                        int* __restrict__ fill, int* __restrict__ rowTok,
                        float* __restrict__ rowW, int* __restrict__ slotRow) {
  int t = blockIdx.x * 256 + threadIdx.x;
  if (t < SEQ) {
#pragma unroll
    for (int j = 0; j < TOPK; j++) {
      int e = topIdx[t * 2 + j];
      int pos = atomicAdd(&fill[e], 1);
      rowTok[pos] = t;
      rowW[pos] = topW[t * 2 + j];
      slotRow[t * 2 + j] = pos;
    }
  }
}

// act = silu(g) * u * routing_weight, stored bf16
__global__ __launch_bounds__(256)
void moe_act_kernel(const float* __restrict__ g, const float* __restrict__ u,
                    const float* __restrict__ rowW, const int* __restrict__ rowTok,
                    unsigned short* __restrict__ actb) {
  long long i = (long long)blockIdx.x * 256 + threadIdx.x;
  if (i < (long long)MAXR * HID) {
    int r = (int)(i / HID);
    float w = (rowTok[r] >= 0) ? rowW[r] : 0.f;
    float gg = g[i];
    float s = gg / (1.f + __expf(-gg));
    actb[i] = f2bf(s * u[i] * w);
  }
}

__global__ __launch_bounds__(256)
void moe_combine_kernel(const float* __restrict__ yRows, const int* __restrict__ slotRow,
                        float* __restrict__ h) {
  int t = blockIdx.x;
  int r0 = slotRow[t * 2], r1 = slotRow[t * 2 + 1];
  for (int d = threadIdx.x; d < DIM; d += 256)
    h[(long long)t * DIM + d] +=
        yRows[(long long)r0 * DIM + d] + yRows[(long long)r1 * DIM + d];
}

// ---------------------------------------------------------------------------
// Host orchestration
// ---------------------------------------------------------------------------
static inline void launch_gemm(const unsigned short* A, const float* W, float* C,
                               unsigned short* Cb, const float* R,
                               int M, int N, int Kd,
                               const int* rowTok, const int* tileExpert,
                               long long wStride, int gridY, hipStream_t s) {
  dim3 g((N + BN - 1) / BN, gridY);
  gemm_bf16_kernel<<<g, 256, 0, s>>>(A, W, C, Cb, R, M, N, Kd, rowTok, tileExpert, wStride);
}

extern "C" void kernel_launch(void* const* d_in, const int* in_sizes, int n_in,
                              void* d_out, int out_size, void* d_ws, size_t ws_size,
                              hipStream_t stream) {
  (void)in_sizes; (void)n_in; (void)out_size; (void)ws_size;
  const int*   tokens    = (const int*)d_in[0];
  const int*   start_pos = (const int*)d_in[1];
  const float* tok_emb   = (const float*)d_in[2];
  const float* wq        = (const float*)d_in[3];
  const float* wk        = (const float*)d_in[4];
  const float* wv        = (const float*)d_in[5];
  const float* wo        = (const float*)d_in[6];
  const float* attn_nw   = (const float*)d_in[7];
  const float* ffn_nw    = (const float*)d_in[8];
  const float* router_w  = (const float*)d_in[9];
  const float* w1        = (const float*)d_in[10];
  const float* w2        = (const float*)d_in[11];
  const float* w3        = (const float*)d_in[12];
  const float* final_nw  = (const float*)d_in[13];
  const float* out_w     = (const float*)d_in[14];
  float* logits = (float*)d_out;

  // workspace carve-out
  char* base = (char*)d_ws;
  size_t off = 0;
  auto alloc = [&](size_t bytes) -> char* {
    char* p = base + off;
    off = (off + bytes + 255) & ~(size_t)255;
    return p;
  };
  float*          h     = (float*)         alloc((size_t)SEQ * DIM * 4);
  unsigned short* xb    = (unsigned short*)alloc((size_t)SEQ * DIM * 2);
  float*          qf    = (float*)         alloc((size_t)SEQ * DIM * 4);
  float*          kf    = (float*)         alloc((size_t)SEQ * DIM * 4);
  unsigned short* qbb   = (unsigned short*)alloc((size_t)SEQ * DIM * 2);
  unsigned short* kbb   = (unsigned short*)alloc((size_t)SEQ * DIM * 2);
  unsigned short* vbb   = (unsigned short*)alloc((size_t)SEQ * DIM * 2);
  unsigned short* obb   = (unsigned short*)alloc((size_t)SEQ * DIM * 2);
  float*          gRows = (float*)         alloc((size_t)MAXR * HID * 4);
  float*          uRows = (float*)         alloc((size_t)MAXR * HID * 4);
  unsigned short* actb  = (unsigned short*)alloc((size_t)MAXR * HID * 2);
  float*          yRows = (float*)         alloc((size_t)MAXR * DIM * 4);
  int*   cnt     = (int*)  alloc(NE * 4);
  int*   fill    = (int*)  alloc(NE * 4);
  int*   tileExp = (int*)  alloc(MAX_TILES * 4);
  int*   rowTok  = (int*)  alloc(MAXR * 4);
  float* rowW    = (float*)alloc(MAXR * 4);
  int*   slotRow = (int*)  alloc((size_t)SEQ * TOPK * 4);
  int*   topIdx  = (int*)  alloc((size_t)SEQ * TOPK * 4);
  float* topW    = (float*)alloc((size_t)SEQ * TOPK * 4);

  embed_kernel<<<SEQ, 256, 0, stream>>>(tokens, tok_emb, h);

  for (int l = 0; l < NL; l++) {
    const float* wq_l = wq + (size_t)l * DIM * DIM;
    const float* wk_l = wk + (size_t)l * DIM * DIM;
    const float* wv_l = wv + (size_t)l * DIM * DIM;
    const float* wo_l = wo + (size_t)l * DIM * DIM;

    // --- attention block ---
    rmsnorm_kernel<<<SEQ, 256, 0, stream>>>(h, attn_nw + (size_t)l * DIM, xb);
    launch_gemm(xb, wq_l, qf, nullptr, nullptr, SEQ, DIM, DIM, nullptr, nullptr, 0, SEQ / BM, stream);
    launch_gemm(xb, wk_l, kf, nullptr, nullptr, SEQ, DIM, DIM, nullptr, nullptr, 0, SEQ / BM, stream);
    launch_gemm(xb, wv_l, nullptr, vbb, nullptr, SEQ, DIM, DIM, nullptr, nullptr, 0, SEQ / BM, stream);
    rope_kernel<<<SEQ, 256, 0, stream>>>(qf, kf, qbb, kbb, start_pos);
    attn_kernel<<<dim3(SEQ / 16, NH), 32, 0, stream>>>(qbb, kbb, vbb, obb);
    launch_gemm(obb, wo_l, h, nullptr, h, SEQ, DIM, DIM, nullptr, nullptr, 0, SEQ / BM, stream);

    // --- MoE block (top-2 grouped GEMM) ---
    rmsnorm_kernel<<<SEQ, 256, 0, stream>>>(h, ffn_nw + (size_t)l * DIM, xb);
    moe_init_kernel<<<(MAXR + 255) / 256, 256, 0, stream>>>(cnt, tileExp, rowTok);
    router_kernel<<<SEQ, 32, 0, stream>>>(xb, router_w + (size_t)l * DIM * NE, topIdx, topW, cnt);
    moe_offsets_kernel<<<1, 1, 0, stream>>>(cnt, fill, tileExp);
    moe_scatter_kernel<<<SEQ / 256, 256, 0, stream>>>(topIdx, topW, fill, rowTok, rowW, slotRow);
    launch_gemm(xb, w1 + (size_t)l * NE * DIM * HID, gRows, nullptr, nullptr,
                MAXR, HID, DIM, rowTok, tileExp, (long long)DIM * HID, MAX_TILES, stream);
    launch_gemm(xb, w3 + (size_t)l * NE * DIM * HID, uRows, nullptr, nullptr,
                MAXR, HID, DIM, rowTok, tileExp, (long long)DIM * HID, MAX_TILES, stream);
    moe_act_kernel<<<(int)(((long long)MAXR * HID + 255) / 256), 256, 0, stream>>>(
        gRows, uRows, rowW, rowTok, actb);
    launch_gemm(actb, w2 + (size_t)l * NE * HID * DIM, yRows, nullptr, nullptr,
                MAXR, DIM, HID, nullptr, tileExp, (long long)HID * DIM, MAX_TILES, stream);
    moe_combine_kernel<<<SEQ, 256, 0, stream>>>(yRows, slotRow, h);
  }

  // --- final norm + vocab projection ---
  rmsnorm_kernel<<<SEQ, 256, 0, stream>>>(h, final_nw, xb);
  launch_gemm(xb, out_w, logits, nullptr, nullptr, SEQ, VOCAB, DIM,
              nullptr, nullptr, 0, SEQ / BM, stream);
}
